// CRKTLayer_24661702214064
// MI455X (gfx1250) — compile-verified, matrix operand order checked by reading, not run on hardware
//
#include <hip/hip_runtime.h>
#include <hip/hip_bf16.h>

// CRKT temporal-decay attention, fused for MI455X (gfx1250, wave32, WMMA).
// B=8, S=1024, DIM=512, H=8, DK=64.

#define Bc 8
#define Sc 1024
#define DIMc 512
#define Hc 8
#define DKc 64

typedef __attribute__((ext_vector_type(16))) __bf16 v16bf;
typedef __attribute__((ext_vector_type(8)))  float  v8f;

#if __has_builtin(__builtin_amdgcn_global_load_async_to_lds_b128)
#define USE_ASYNC_V 1
// exact pointee type from the builtin's signature (v4i = vector_size(16) int)
typedef int async_v4i __attribute__((vector_size(16)));
#define GAS_PTR(p) ((__attribute__((address_space(1))) async_v4i*)(p))
#define LAS_PTR(p) ((__attribute__((address_space(3))) async_v4i*)(p))
#endif

// ---------------------------------------------------------------------------
// Fragment loaders.  CDNA5 16-bit A-fragment (16x32, MxK): lane L (r=L&15,
// half=L>>4) holds K = half*8 .. half*8+7  and  K = 16+half*8 .. 16+half*8+7
// of row r.  For D = A*B^T with both row-major, the B-fragment loads with the
// exact same pattern (lane indexes the N row of B^T).
// ---------------------------------------------------------------------------
__device__ __forceinline__ v16bf frag_from_f32(const float* p0, int ld, int lane) {
  const int r = lane & 15, hf = lane >> 4;
  const float* p = p0 + r * ld + hf * 8;
  v16bf f;
#pragma unroll
  for (int i = 0; i < 8; ++i) f[i] = (__bf16)p[i];
#pragma unroll
  for (int i = 0; i < 8; ++i) f[8 + i] = (__bf16)p[16 + i];
  return f;
}

__device__ __forceinline__ v16bf frag_from_bf16(const __bf16* p0, int ld, int lane) {
  const int r = lane & 15, hf = lane >> 4;
  const __bf16* p = p0 + r * ld + hf * 8;
  v16bf f;
#pragma unroll
  for (int i = 0; i < 8; ++i) f[i] = p[i];
#pragma unroll
  for (int i = 0; i < 8; ++i) f[8 + i] = p[16 + i];
  return f;
}

// B-fragment for alpha@V gathered from a row-major LDS tile vrow[t][64]:
// lane L holds column d, VGPR v holds K(t) pairs in the A-pattern order.
__device__ __forceinline__ v16bf frag_b_from_rows(const __bf16* vrow, int d, int lane) {
  const int hf = lane >> 4;
  v16bf f;
#pragma unroll
  for (int i = 0; i < 8; ++i) f[i] = vrow[(hf * 8 + i) * DKc + d];
#pragma unroll
  for (int i = 0; i < 8; ++i) f[8 + i] = vrow[(16 + hf * 8 + i) * DKc + d];
  return f;
}

__device__ __forceinline__ v8f wmma_bf16(v16bf a, v16bf b, v8f c) {
  // v_wmma_f32_16x16x32_bf16  (8-arg form, f32 accumulate)
  return __builtin_amdgcn_wmma_f32_16x16x32_bf16(
      /*neg_a=*/false, a, /*neg_b=*/false, b,
      /*c_mod=*/(short)0, c, /*reuse_a=*/false, /*reuse_b=*/false);
}

// Reductions within the 16-lane row group (C/D layout: one matrix row lives
// in lanes [0..15] or [16..31] of the wave32).
__device__ __forceinline__ float rowmax16(float x) {
#pragma unroll
  for (int off = 1; off < 16; off <<= 1) x = fmaxf(x, __shfl_xor(x, off, 16));
  return x;
}
__device__ __forceinline__ float rowsum16(float x) {
#pragma unroll
  for (int off = 1; off < 16; off <<= 1) x += __shfl_xor(x, off, 16);
  return x;
}

#ifdef USE_ASYNC_V
__device__ __forceinline__ void wait_async0() {
#if __has_builtin(__builtin_amdgcn_s_wait_asynccnt)
  __builtin_amdgcn_s_wait_asynccnt(0);
#else
  asm volatile("s_wait_asynccnt 0" ::: "memory");
#endif
  asm volatile("" ::: "memory");
}
#endif

// ---------------------------------------------------------------------------
// Kernel 0: f32 -> bf16 weight conversion (run once per weight matrix).
// ---------------------------------------------------------------------------
__global__ __launch_bounds__(256) void cvt_kernel(const float* __restrict__ src,
                                                  __bf16* __restrict__ dst) {
  const int i = (blockIdx.x * 256 + threadIdx.x) * 4;
#pragma unroll
  for (int c = 0; c < 4; ++c) dst[i + c] = (__bf16)src[i + c];
}

// ---------------------------------------------------------------------------
// Kernel 1: Y = X @ W^T + bias, output stored bf16 head-major [B*H, S, DK].
// Block = 256 threads = 8 waves; each wave computes a 16x64 output tile.
// Grid: (8192/128, 512/64) = (64, 8).  W is pre-converted bf16.
// ---------------------------------------------------------------------------
__global__ __launch_bounds__(256) void proj_kernel(
    const float* __restrict__ X, const __bf16* __restrict__ W,
    const float* __restrict__ bias, __bf16* __restrict__ Y) {
  const int lane = threadIdx.x & 31, wv = threadIdx.x >> 5;
  const int m0 = blockIdx.x * 128 + wv * 16;
  const int n0 = blockIdx.y * 64;
  v8f acc[4] = {};
  for (int k = 0; k < DIMc; k += 32) {
    v16bf af = frag_from_f32(X + (size_t)m0 * DIMc + k, DIMc, lane);
#pragma unroll
    for (int ct = 0; ct < 4; ++ct) {
      v16bf bf = frag_from_bf16(W + (size_t)(n0 + ct * 16) * DIMc + k, DIMc, lane);
      acc[ct] = wmma_bf16(af, bf, acc[ct]);
    }
  }
  const int hf = lane >> 4, nc = lane & 15;
#pragma unroll
  for (int ct = 0; ct < 4; ++ct) {
    const int n = n0 + ct * 16 + nc;
    const float bb = bias[n];
    const int h = n >> 6, d = n & 63;
#pragma unroll
    for (int i = 0; i < 8; ++i) {
      const int m = m0 + hf * 8 + i;
      const int b = m >> 10, s = m & 1023;
      Y[((size_t)(b * Hc + h) * Sc + s) * DKc + d] = (__bf16)(acc[ct][i] + bb);
    }
  }
}

// ---------------------------------------------------------------------------
// Kernel 2: fused causal attention with temporal decay.
// Grid (B*H=64, S/128=8); 8 waves/block, each wave owns one 16-row Q tile.
// Pass 1: online softmax-1 stats (m1, l1).
// Pass 2: reconstruct masked_attn, lane-prefix cumsum, decay-rescaled scores,
//         online softmax-2 fused with alpha@V WMMA accumulation.
// K/V per head are 128 KB bf16 -> fully L2-resident (192 MB L2).
// ---------------------------------------------------------------------------
__global__ __launch_bounds__(256) void attn_kernel(
    const __bf16* __restrict__ Q, const __bf16* __restrict__ K,
    const __bf16* __restrict__ V, const float* __restrict__ decay,
    __bf16* __restrict__ O) {
  __shared__ __bf16 pst_s[8][16 * 32];   // per-wave alpha staging (A layout)
#ifdef USE_ASYNC_V
  __shared__ __bf16 vrow_s[8][32 * DKc]; // per-wave raw V tile [t][d] (async)
#else
  __shared__ __bf16 vt_s[8][16 * 32];    // per-wave transposed V tile [d][t]
#endif
  const int lane = threadIdx.x & 31, wv = threadIdx.x >> 5;
  __bf16* pst = pst_s[wv];
  const int bh = blockIdx.x;
  const int s0 = blockIdx.y * 128 + wv * 16;
  const float dec = fabsf(decay[bh & (Hc - 1)]);
  const size_t base = (size_t)bh * Sc * DKc;
  const int hf = lane >> 4, nc = lane & 15;
  const int tmax = s0 + 15;

  const v16bf qf0 = frag_from_bf16(Q + base + (size_t)s0 * DKc + 0,  DKc, lane);
  const v16bf qf1 = frag_from_bf16(Q + base + (size_t)s0 * DKc + 32, DKc, lane);

  float m1[8], l1[8];
#pragma unroll
  for (int i = 0; i < 8; ++i) { m1[i] = -3.0e38f; l1[i] = 0.f; }

  // ---- Pass 1: softmax-1 statistics --------------------------------------
  for (int t0 = 0; t0 <= tmax; t0 += 16) {
    if (t0 + 16 <= tmax)  // prefetch next K tile (-> global_prefetch_b8)
      __builtin_prefetch(K + base + (size_t)(t0 + 16 + nc) * DKc, 0, 1);
    v16bf kf0 = frag_from_bf16(K + base + (size_t)t0 * DKc + 0,  DKc, lane);
    v16bf kf1 = frag_from_bf16(K + base + (size_t)t0 * DKc + 32, DKc, lane);
    v8f c = {};
    c = wmma_bf16(qf0, kf0, c);
    c = wmma_bf16(qf1, kf1, c);
    const int j = t0 + nc;
#pragma unroll
    for (int i = 0; i < 8; ++i) {
      const int row = s0 + hf * 8 + i;
      const bool valid = (j <= row);
      float sc = valid ? c[i] * 0.125f : -3.0e38f;  // 1/sqrt(64)
      float mnew = fmaxf(m1[i], rowmax16(sc));
      float p = valid ? __expf(sc - mnew) : 0.f;
      l1[i] = l1[i] * __expf(m1[i] - mnew) + rowsum16(p);
      m1[i] = mnew;
    }
  }

  // ---- Pass 2: cumsum + decay + softmax-2 + alpha@V -----------------------
  float m2[8], l2[8], cum[8];
#pragma unroll
  for (int i = 0; i < 8; ++i) { m2[i] = -3.0e38f; l2[i] = 0.f; cum[i] = 0.f; }
  v8f acc[4] = {};

  for (int t0 = 0; t0 <= tmax; t0 += 32) {
#ifdef USE_ASYNC_V
    // Kick off the V tile DMA early: each lane pulls its row t0+lane of V
    // (64 bf16 = 8 x B128) straight into LDS, bypassing VGPRs (ASYNCcnt).
    __bf16* vrow = vrow_s[wv];
    {
      asm volatile("" ::: "memory");
      const __bf16* gsrc = V + base + (size_t)(t0 + lane) * DKc;
#pragma unroll
      for (int c = 0; c < 8; ++c)
        __builtin_amdgcn_global_load_async_to_lds_b128(
            GAS_PTR(gsrc + c * 8), LAS_PTR(vrow + lane * DKc + c * 8), 0, 0);
    }
#endif
    float s2[2][8];
#pragma unroll
    for (int sub = 0; sub < 2; ++sub) {
      const int tt = t0 + sub * 16;
      v16bf kf0 = frag_from_bf16(K + base + (size_t)tt * DKc + 0,  DKc, lane);
      v16bf kf1 = frag_from_bf16(K + base + (size_t)tt * DKc + 32, DKc, lane);
      v8f c = {};
      c = wmma_bf16(qf0, kf0, c);
      c = wmma_bf16(qf1, kf1, c);
      const int j = tt + nc;
#pragma unroll
      for (int i = 0; i < 8; ++i) {
        const int row = s0 + hf * 8 + i;
        const bool valid = (j <= row);
        float sc = valid ? c[i] * 0.125f : 0.f;
        float p1 = valid ? __expf(sc - m1[i]) / l1[i] : 0.f;  // masked_attn
        // inclusive prefix sum across the 16-lane row group (column order)
        float x = p1;
#pragma unroll
        for (int off = 1; off < 16; off <<= 1) {
          float y = __shfl_up(x, off, 16);
          if (nc >= off) x += y;
        }
        const float cinc = cum[i] + x;          // inclusive cumsum at col j
        cum[i] += __shfl(x, 15, 16);            // row total of this sub-tile
        const float td   = (float)(row - j);    // |j-i| on causal side
        const float dist = fmaxf(td * (1.f - cinc), 0.f);
        s2[sub][i] = valid ? sc * __expf(-dec * dist) : -3.0e38f;
      }
    }
    // online softmax-2 across the 32 columns of this chunk
    float pv[2][8];
#pragma unroll
    for (int i = 0; i < 8; ++i) {
      float mt = fmaxf(rowmax16(s2[0][i]), rowmax16(s2[1][i]));
      float mnew  = fmaxf(m2[i], mt);
      float alpha = __expf(m2[i] - mnew);
      float p0 = (s2[0][i] > -1.0e38f) ? __expf(s2[0][i] - mnew) : 0.f;
      float p1 = (s2[1][i] > -1.0e38f) ? __expf(s2[1][i] - mnew) : 0.f;
      l2[i] = l2[i] * alpha + rowsum16(p0) + rowsum16(p1);
      m2[i] = mnew;
      pv[0][i] = p0; pv[1][i] = p1;
#pragma unroll
      for (int ct = 0; ct < 4; ++ct) acc[ct][i] *= alpha;
    }
    // stage alpha tile (D layout -> A layout) through per-wave LDS;
    // same-wave DS ops are in-order, no barrier required.
#pragma unroll
    for (int sub = 0; sub < 2; ++sub)
#pragma unroll
      for (int i = 0; i < 8; ++i)
        pst[(hf * 8 + i) * 32 + sub * 16 + nc] = (__bf16)pv[sub][i];
    const v16bf pa = frag_from_bf16(pst, 32, lane);
#ifdef USE_ASYNC_V
    wait_async0();  // s_wait_asynccnt 0: V tile resident in LDS
#pragma unroll
    for (int ct = 0; ct < 4; ++ct) {
      const v16bf vb = frag_b_from_rows(vrow_s[wv], ct * 16 + nc, lane);
      acc[ct] = wmma_bf16(pa, vb, acc[ct]);
    }
#else
    // fallback: stage transposed 16x32 tile VT[d][t] per 16-wide d block
    __bf16* vt = vt_s[wv];
#pragma unroll
    for (int ct = 0; ct < 4; ++ct) {
      const __bf16* vsrc = V + base + (size_t)(t0 + lane) * DKc + ct * 16;
#pragma unroll
      for (int dd = 0; dd < 16; ++dd) vt[dd * 32 + lane] = vsrc[dd];
      const v16bf vb = frag_from_bf16(vt, 32, lane);
      acc[ct] = wmma_bf16(pa, vb, acc[ct]);
    }
#endif
  }

  // normalize and emit bf16 head-major output
#pragma unroll
  for (int ct = 0; ct < 4; ++ct)
#pragma unroll
    for (int i = 0; i < 8; ++i) {
      const int row = s0 + hf * 8 + i;
      O[base + (size_t)row * DKc + ct * 16 + nc] = (__bf16)(acc[ct][i] / l2[i]);
    }
}

// ---------------------------------------------------------------------------
// Kernel 3: out = LN(concat(A) @ Wo^T + bo) * gamma + beta.
// One block per 16-row slab; 8 waves each own a 64-col span (4 WMMA tiles),
// results staged in LDS for the full-row LayerNorm.  Wo pre-converted bf16.
// ---------------------------------------------------------------------------
__global__ __launch_bounds__(256) void outproj_ln_kernel(
    const __bf16* __restrict__ A, const __bf16* __restrict__ Wo,
    const float* __restrict__ bo, const float* __restrict__ gamma,
    const float* __restrict__ beta, float* __restrict__ out) {
  __shared__ float tile[16 * DIMc];  // 32 KB
  const int lane = threadIdx.x & 31, wv = threadIdx.x >> 5;
  const int m0 = blockIdx.x * 16;
  const int b = m0 >> 10, sl0 = m0 & 1023;
  const int n0 = wv * 64;
  v8f acc[4] = {};
  for (int k = 0; k < DIMc; k += 32) {
    const int hd = k >> 6, inner = k & 63;  // head-major gather of concat(A)
    const __bf16* ap = A + ((size_t)(b * Hc + hd) * Sc + sl0) * DKc + inner;
    v16bf af = frag_from_bf16(ap, DKc, lane);
#pragma unroll
    for (int ct = 0; ct < 4; ++ct) {
      v16bf bf = frag_from_bf16(Wo + (size_t)(n0 + ct * 16) * DIMc + k, DIMc, lane);
      acc[ct] = wmma_bf16(af, bf, acc[ct]);
    }
  }
  const int hf = lane >> 4, nc = lane & 15;
#pragma unroll
  for (int ct = 0; ct < 4; ++ct) {
    const int n = n0 + ct * 16 + nc;
    const float bb = bo[n];
#pragma unroll
    for (int i = 0; i < 8; ++i) tile[(hf * 8 + i) * DIMc + n] = acc[ct][i] + bb;
  }
  __syncthreads();
  // LayerNorm: 16 threads per row (lanes of one 16-group share a row)
  const int row = threadIdx.x >> 4, sub = threadIdx.x & 15;
  const float* trow = tile + row * DIMc;
  float sum = 0.f, sumsq = 0.f;
#pragma unroll
  for (int c = 0; c < 32; ++c) {
    float x = trow[sub * 32 + c];
    sum += x; sumsq += x * x;
  }
  sum = rowsum16(sum); sumsq = rowsum16(sumsq);
  const float mu  = sum * (1.f / DIMc);
  const float var = sumsq * (1.f / DIMc) - mu * mu;
  const float rstd = rsqrtf(var + 1e-5f);
  float* orow = out + (size_t)(m0 + row) * DIMc;
#pragma unroll
  for (int c = 0; c < 32; ++c) {
    const int col = sub * 32 + c;
    orow[col] = (trow[col] - mu) * rstd * gamma[col] + beta[col];
  }
}

// ---------------------------------------------------------------------------
extern "C" void kernel_launch(void* const* d_in, const int* in_sizes, int n_in,
                              void* d_out, int out_size, void* d_ws, size_t ws_size,
                              hipStream_t stream) {
  const float* query  = (const float*)d_in[0];
  const float* key_in = (const float*)d_in[1];
  const float* value  = (const float*)d_in[2];
  const float* Wq     = (const float*)d_in[3];
  const float* bq     = (const float*)d_in[4];
  const float* Wv     = (const float*)d_in[5];
  const float* bv     = (const float*)d_in[6];
  const float* Wo     = (const float*)d_in[7];
  const float* bo     = (const float*)d_in[8];
  const float* decay  = (const float*)d_in[9];
  const float* gamma  = (const float*)d_in[10];
  const float* beta   = (const float*)d_in[11];
  float* out = (float*)d_out;

  const size_t NE = (size_t)Bc * Sc * DIMc;  // 4.19M elements
  const size_t NW = (size_t)DIMc * DIMc;     // 262144 elements
  __bf16* qb  = (__bf16*)d_ws;
  __bf16* kb  = qb + NE;
  __bf16* vb  = kb + NE;
  __bf16* ab  = vb + NE;
  __bf16* wqb = ab + NE;
  __bf16* wvb = wqb + NW;
  __bf16* wob = wvb + NW;   // total ~33.5 MiB of workspace

  // one-time (per launch) weight conversion: 3 x 512x512
  cvt_kernel<<<NW / 1024, 256, 0, stream>>>(Wq, wqb);
  cvt_kernel<<<NW / 1024, 256, 0, stream>>>(Wv, wvb);
  cvt_kernel<<<NW / 1024, 256, 0, stream>>>(Wo, wob);

  dim3 gproj(64, 8);
  proj_kernel<<<gproj, 256, 0, stream>>>(query,  wqb, bq, qb);
  proj_kernel<<<gproj, 256, 0, stream>>>(key_in, wqb, bq, kb);  // kq_same
  proj_kernel<<<gproj, 256, 0, stream>>>(value,  wvb, bv, vb);

  dim3 gattn(Bc * Hc, Sc / 128);
  attn_kernel<<<gattn, 256, 0, stream>>>(qb, kb, vb, decay, ab);

  outproj_ln_kernel<<<Bc * Sc / 16, 256, 0, stream>>>(ab, wob, bo, gamma, beta, out);
}
